// DeepSeekMoE_43731357008651
// MI455X (gfx1250) — matrix operand order repaired and checked by reading, not compiled, not run on hardware
//
#include <hip/hip_runtime.h>
#include <hip/hip_bf16.h>
#include <math.h>

// Problem dims (fixed by the reference)
#define T_TOK 4096
#define DMODEL 1024
#define NEXP 8
#define TOPK 2
#define ISZ 512
#define SISZ 2048
// expert segments padded to 128 rows (one workgroup super-tile):
// worst case sum ceil128(n_e) <= 8192 + 8*127 = 9208 -> 9216
#define CAP 9216
#define MSUPER_MAX (CAP / 128)   // 72

typedef __attribute__((ext_vector_type(16))) __bf16 v16bf;
typedef __attribute__((ext_vector_type(8)))  __bf16 v8bf;
typedef __attribute__((ext_vector_type(8)))  float  v8f;
typedef unsigned short u16;
typedef unsigned int   u32;

__device__ __forceinline__ u16 f2bf(float f) {
  u32 u = __float_as_uint(f);
  u32 r = u + 0x7FFFu + ((u >> 16) & 1u);   // round-to-nearest-even
  return (u16)(r >> 16);
}

__device__ __forceinline__ v8f wmma_bf16(v16bf a, v16bf b, v8f c) {
  return __builtin_amdgcn_wmma_f32_16x16x32_bf16(false, a, false, b, (short)0, c,
                                                 false, false);
}

// A-fragment: 16x32 bf16 tile, row = lane&15.
// lanes 0-15: K = 0..15; lanes 16-31: K = 16..31 (b=8 -> offset 2b).
__device__ __forceinline__ v16bf load_a(const u16* __restrict__ arow, int kbase, int b) {
  v16bf a;
  a = *(const v16bf*)(arow + kbase + 2 * b);
  return a;
}

// ---------------- conversion / packing ----------------

__global__ void k_f32_to_bf16(const float* __restrict__ src, u16* __restrict__ dst, int n) {
  int i = blockIdx.x * blockDim.x + threadIdx.x;
  if (i < n) dst[i] = f2bf(src[i]);
}

// Pack a row-major [Kd x Nd] fp32 matrix (nslices of them) into WMMA-B tiles:
// tile (kt,nt) = 32x16 block, 512 halves; element (kk,nn):
//   lane = nn + 16*(kk>>4), half = kk&15; lane's 16 halves are contiguous.
__global__ void k_pack_b(const float* __restrict__ src, u16* __restrict__ dst,
                         int Kd, int Nd, int nslices) {
  int per = Kd * Nd;
  long long i = (long long)blockIdx.x * blockDim.x + threadIdx.x;
  if (i >= (long long)per * nslices) return;
  int s   = (int)(i / per);
  int rem = (int)(i - (long long)s * per);
  int k = rem / Nd, n = rem % Nd;
  int kt = k >> 5, kk = k & 31;
  int nt = n >> 4, nn = n & 15;
  int lane = nn + ((kk >> 4) << 4);
  int h    = kk & 15;
  int off  = ((kt * (Nd >> 4) + nt) << 9) + lane * 16 + h;
  dst[(long long)s * per + off] = f2bf(src[i]);
}

// ---------------- router ----------------

__global__ void k_router(const float* __restrict__ x, const float* __restrict__ Wr,
                         float* __restrict__ logits_out, int* __restrict__ topi,
                         float* __restrict__ topw) {
  int wave = (blockIdx.x * blockDim.x + threadIdx.x) >> 5;
  int lane = threadIdx.x & 31;
  if (wave >= T_TOK) return;
  const float* xr = x + (long long)wave * DMODEL;
  float acc[NEXP];
#pragma unroll
  for (int e = 0; e < NEXP; ++e) acc[e] = 0.f;
  for (int d = lane; d < DMODEL; d += 32) {
    float xv = xr[d];
    const float* wr = Wr + d * NEXP;
#pragma unroll
    for (int e = 0; e < NEXP; ++e) acc[e] += xv * wr[e];
  }
#pragma unroll
  for (int e = 0; e < NEXP; ++e) {
#pragma unroll
    for (int m = 16; m >= 1; m >>= 1) acc[e] += __shfl_xor(acc[e], m, 32);
  }
  if (lane == 0) {
    int i0 = 0; float v0 = acc[0];
#pragma unroll
    for (int e = 1; e < NEXP; ++e) if (acc[e] > v0) { v0 = acc[e]; i0 = e; }
    int i1 = (i0 == 0) ? 1 : 0; float v1 = acc[i1];
#pragma unroll
    for (int e = 0; e < NEXP; ++e)
      if (e != i0 && acc[e] > v1) { v1 = acc[e]; i1 = e; }
    float w0 = 1.f / (1.f + __expf(v1 - v0));   // softmax over {v0,v1}
    topi[wave * 2]     = i0;  topi[wave * 2 + 1] = i1;
    topw[wave * 2]     = w0;
    topw[wave * 2 + 1] = 1.f - w0;
#pragma unroll
    for (int e = 0; e < NEXP; ++e)
      logits_out[(long long)wave * NEXP + e] = acc[e];
  }
}

// meta layout (ints): [0..8] padded_off[0..E] (128-aligned), [9..16] scatter cursors
__global__ void k_count(const int* __restrict__ topi, int* __restrict__ meta,
                        int* __restrict__ token_list, float* __restrict__ slot_w) {
  __shared__ int cnt[NEXP];
  if (threadIdx.x < NEXP) cnt[threadIdx.x] = 0;
  __syncthreads();
  for (int i = threadIdx.x; i < T_TOK * TOPK; i += blockDim.x)
    atomicAdd(&cnt[topi[i]], 1);
  __syncthreads();
  if (threadIdx.x == 0) {
    int off = 0;
    for (int e = 0; e < NEXP; ++e) {
      meta[e]     = off;
      meta[9 + e] = off;
      off += (cnt[e] + 127) & ~127;
    }
    meta[NEXP] = off;  // total padded slots (multiple of 128)
  }
  for (int i = threadIdx.x; i < CAP; i += blockDim.x) {
    token_list[i] = -1;
    slot_w[i] = 0.f;
  }
}

__global__ void k_scatter(const int* __restrict__ topi, const float* __restrict__ topw,
                          int* __restrict__ meta, int* __restrict__ token_list,
                          float* __restrict__ slot_w, int* __restrict__ slot_of) {
  int t = blockIdx.x * blockDim.x + threadIdx.x;
  if (t >= T_TOK) return;
#pragma unroll
  for (int k = 0; k < TOPK; ++k) {
    int e = topi[t * 2 + k];
    int slot = atomicAdd(&meta[9 + e], 1);
    token_list[slot] = t;
    slot_w[slot]     = topw[t * 2 + k];
    slot_of[t * 2 + k] = slot;
  }
}

// ---------------- routed expert GEMMs ----------------
// Workgroup = 8 waves = 128-row super-tile; B tiles staged in LDS
// (double-buffered, software-pipelined, one barrier per K-step) and shared
// by all 8 waves. Gate+up kernels use 32-col groups (4 accumulators/wave)
// to stay within VGPR budget; down kernels use 64-col groups.

// gate+up: 4 B-tiles/kt (2 gate + 2 up) = 4KB staged; fused SiLU*up*gate_w epilogue.
__global__ __launch_bounds__(256)
void k_moe_gateup(const u16* __restrict__ xb,
                  const u16* __restrict__ Wg_pk, const u16* __restrict__ Wu_pk,
                  const int* __restrict__ meta,
                  const int* __restrict__ token_list,
                  const float* __restrict__ slot_w,
                  u16* __restrict__ hbuf) {
  __shared__ __align__(64) u16 sb[2][4 * 512];      // 8 KB
  const int NG = ISZ / 32;                          // 16 column groups of 32
  int ng    = blockIdx.x & (NG - 1);
  int super = blockIdx.x / NG;
  int padded_total = meta[NEXP];
  int sm_base = super * 128;
  if (sm_base >= padded_total) return;
  int e = 0;
#pragma unroll
  for (int i = 1; i < NEXP; ++i) if (sm_base >= meta[i]) e = i;
  const u16* Wg_e = Wg_pk + (long long)e * DMODEL * ISZ;
  const u16* Wu_e = Wu_pk + (long long)e * DMODEL * ISZ;

  int tid  = threadIdx.x;
  int lane = tid & 31;
  int wv   = tid >> 5;
  int m_base = sm_base + wv * 16;
  int tok = token_list[m_base + (lane & 15)];
  const u16* arow = xb + (long long)((tok < 0) ? 0 : tok) * DMODEL;
  int b = (lane >> 4) << 3;

  // staging: thread stages 16B; tile g = tid>>6 (0-1 gate, 2-3 up), chunk c = tid&63
  int g = tid >> 6, c = tid & 63;
  const u16* gsrc = (g < 2) ? Wg_e : Wu_e;
  int jt = g & 1;
  const int NT16 = ISZ >> 4;                        // 32
  const int NKT  = DMODEL / 32;                     // 32
  *(v8bf*)&sb[0][g * 512 + c * 8] =
      *(const v8bf*)(gsrc + ((ng * 2 + jt) << 9) + c * 8);
  __syncthreads();

  v8f accg[2] = {}, accu[2] = {};
  for (int kt = 0; kt < NKT; ++kt) {
    int cur = kt & 1;
    v8bf st;
    bool more = (kt + 1 < NKT);
    if (more)
      st = *(const v8bf*)(gsrc + (((kt + 1) * NT16 + ng * 2 + jt) << 9) + c * 8);
    v16bf a = load_a(arow, kt * 32, b);
#pragma unroll
    for (int j = 0; j < 2; ++j) {
      v16bf bg = *(const v16bf*)&sb[cur][j * 512 + lane * 16];
      accg[j] = wmma_bf16(a, bg, accg[j]);
      v16bf bu = *(const v16bf*)&sb[cur][(2 + j) * 512 + lane * 16];
      accu[j] = wmma_bf16(a, bu, accu[j]);
    }
    if (more) *(v8bf*)&sb[cur ^ 1][g * 512 + c * 8] = st;
    __syncthreads();
  }

  int mloc = (lane >> 4) << 3;
  float wrow[8];
#pragma unroll
  for (int v = 0; v < 8; ++v) wrow[v] = slot_w[m_base + mloc + v];
  int ncol0 = ng * 32 + (lane & 15);
#pragma unroll
  for (int j = 0; j < 2; ++j) {
    int n = ncol0 + j * 16;
#pragma unroll
    for (int v = 0; v < 8; ++v) {
      float gg = accg[j][v], uu = accu[j][v];
      float hv = (gg / (1.f + __expf(-gg))) * uu * wrow[v];
      hbuf[(long long)(m_base + mloc + v) * ISZ + n] = f2bf(hv);
    }
  }
}

// down: 4 B-tiles/kt = 4KB staged; y fp32 per slot (gate weight folded into h)
__global__ __launch_bounds__(256)
void k_moe_down(const u16* __restrict__ hbuf, const u16* __restrict__ Wd_pk,
                const int* __restrict__ meta, float* __restrict__ ybuf) {
  __shared__ __align__(64) u16 sb[2][4 * 512];      // 8 KB
  const int NG = DMODEL / 64;                       // 16
  int ng    = blockIdx.x & (NG - 1);
  int super = blockIdx.x / NG;
  int padded_total = meta[NEXP];
  int sm_base = super * 128;
  if (sm_base >= padded_total) return;
  int e = 0;
#pragma unroll
  for (int i = 1; i < NEXP; ++i) if (sm_base >= meta[i]) e = i;
  const u16* Wd_e = Wd_pk + (long long)e * ISZ * DMODEL;

  int tid  = threadIdx.x;
  int lane = tid & 31;
  int wv   = tid >> 5;
  int m_base = sm_base + wv * 16;
  const u16* arow = hbuf + (long long)(m_base + (lane & 15)) * ISZ;
  int b = (lane >> 4) << 3;

  int g = tid >> 6, c = tid & 63;                   // tile g (0..3), 16B chunk c
  const int NT16 = DMODEL >> 4;                     // 64
  const int NKT  = ISZ / 32;                        // 16
  *(v8bf*)&sb[0][g * 512 + c * 8] =
      *(const v8bf*)(Wd_e + ((ng * 4 + g) << 9) + c * 8);
  __syncthreads();

  v8f acc[4] = {};
  for (int kt = 0; kt < NKT; ++kt) {
    int cur = kt & 1;
    v8bf st;
    bool more = (kt + 1 < NKT);
    if (more)
      st = *(const v8bf*)(Wd_e + (((kt + 1) * NT16 + ng * 4 + g) << 9) + c * 8);
    v16bf a = load_a(arow, kt * 32, b);
#pragma unroll
    for (int j = 0; j < 4; ++j) {
      v16bf bw = *(const v16bf*)&sb[cur][j * 512 + lane * 16];
      acc[j] = wmma_bf16(a, bw, acc[j]);
    }
    if (more) *(v8bf*)&sb[cur ^ 1][g * 512 + c * 8] = st;
    __syncthreads();
  }

  int mloc  = (lane >> 4) << 3;
  int ncol0 = ng * 64 + (lane & 15);
#pragma unroll
  for (int j = 0; j < 4; ++j)
#pragma unroll
    for (int v = 0; v < 8; ++v)
      ybuf[(long long)(m_base + mloc + v) * DMODEL + ncol0 + j * 16] = acc[j][v];
}

// ---------------- shared expert ----------------

__global__ __launch_bounds__(256)
void k_sh_gateup(const u16* __restrict__ xb,
                 const u16* __restrict__ Wgs_pk, const u16* __restrict__ Wus_pk,
                 u16* __restrict__ hs) {
  __shared__ __align__(64) u16 sb[2][4 * 512];      // 8 KB
  const int NG = SISZ / 32;                         // 64 column groups of 32
  int ng    = blockIdx.x & (NG - 1);
  int super = blockIdx.x / NG;

  int tid  = threadIdx.x;
  int lane = tid & 31;
  int wv   = tid >> 5;
  int m_base = super * 128 + wv * 16;
  const u16* arow = xb + (long long)(m_base + (lane & 15)) * DMODEL;
  int b = (lane >> 4) << 3;

  int g = tid >> 6, c = tid & 63;
  const u16* gsrc = (g < 2) ? Wgs_pk : Wus_pk;
  int jt = g & 1;
  const int NT16 = SISZ >> 4;                       // 128
  const int NKT  = DMODEL / 32;                     // 32
  *(v8bf*)&sb[0][g * 512 + c * 8] =
      *(const v8bf*)(gsrc + ((ng * 2 + jt) << 9) + c * 8);
  __syncthreads();

  v8f accg[2] = {}, accu[2] = {};
  for (int kt = 0; kt < NKT; ++kt) {
    int cur = kt & 1;
    v8bf st;
    bool more = (kt + 1 < NKT);
    if (more)
      st = *(const v8bf*)(gsrc + (((kt + 1) * NT16 + ng * 2 + jt) << 9) + c * 8);
    v16bf a = load_a(arow, kt * 32, b);
#pragma unroll
    for (int j = 0; j < 2; ++j) {
      v16bf bg = *(const v16bf*)&sb[cur][j * 512 + lane * 16];
      accg[j] = wmma_bf16(a, bg, accg[j]);
      v16bf bu = *(const v16bf*)&sb[cur][(2 + j) * 512 + lane * 16];
      accu[j] = wmma_bf16(a, bu, accu[j]);
    }
    if (more) *(v8bf*)&sb[cur ^ 1][g * 512 + c * 8] = st;
    __syncthreads();
  }

  int mloc  = (lane >> 4) << 3;
  int ncol0 = ng * 32 + (lane & 15);
#pragma unroll
  for (int j = 0; j < 2; ++j) {
    int n = ncol0 + j * 16;
#pragma unroll
    for (int v = 0; v < 8; ++v) {
      float gg = accg[j][v], uu = accu[j][v];
      hs[(long long)(m_base + mloc + v) * SISZ + n] = f2bf((gg / (1.f + __expf(-gg))) * uu);
    }
  }
}

// shared down fused with combine: out = shared + y[slot0] + y[slot1]
__global__ __launch_bounds__(256)
void k_sh_down_combine(const u16* __restrict__ hs, const u16* __restrict__ Wds_pk,
                       const float* __restrict__ ybuf,
                       const int* __restrict__ slot_of,
                       float* __restrict__ out) {
  __shared__ __align__(64) u16 sb[2][4 * 512];      // 8 KB
  const int NG = DMODEL / 64;                       // 16
  int ng    = blockIdx.x & (NG - 1);
  int super = blockIdx.x / NG;

  int tid  = threadIdx.x;
  int lane = tid & 31;
  int wv   = tid >> 5;
  int m_base = super * 128 + wv * 16;
  const u16* arow = hs + (long long)(m_base + (lane & 15)) * SISZ;
  int b = (lane >> 4) << 3;

  int g = tid >> 6, c = tid & 63;
  const int NT16 = DMODEL >> 4;                     // 64
  const int NKT  = SISZ / 32;                       // 64
  *(v8bf*)&sb[0][g * 512 + c * 8] =
      *(const v8bf*)(Wds_pk + ((ng * 4 + g) << 9) + c * 8);
  __syncthreads();

  v8f acc[4] = {};
  for (int kt = 0; kt < NKT; ++kt) {
    int cur = kt & 1;
    v8bf st;
    bool more = (kt + 1 < NKT);
    if (more)
      st = *(const v8bf*)(Wds_pk + (((kt + 1) * NT16 + ng * 4 + g) << 9) + c * 8);
    v16bf a = load_a(arow, kt * 32, b);
#pragma unroll
    for (int j = 0; j < 4; ++j) {
      v16bf bw = *(const v16bf*)&sb[cur][j * 512 + lane * 16];
      acc[j] = wmma_bf16(a, bw, acc[j]);
    }
    if (more) *(v8bf*)&sb[cur ^ 1][g * 512 + c * 8] = st;
    __syncthreads();
  }

  int mloc  = (lane >> 4) << 3;
  int ncol0 = ng * 64 + (lane & 15);
#pragma unroll
  for (int v = 0; v < 8; ++v) {
    int t  = m_base + mloc + v;
    int s0 = slot_of[t * 2], s1 = slot_of[t * 2 + 1];
    const float* y0 = ybuf + (long long)s0 * DMODEL;
    const float* y1 = ybuf + (long long)s1 * DMODEL;
#pragma unroll
    for (int j = 0; j < 4; ++j) {
      int n = ncol0 + j * 16;
      out[(long long)t * DMODEL + n] = acc[j][v] + y0[n] + y1[n];
    }
  }
}

// ---------------- launch ----------------

extern "C" void kernel_launch(void* const* d_in, const int* in_sizes, int n_in,
                              void* d_out, int out_size, void* d_ws, size_t ws_size,
                              hipStream_t stream) {
  (void)in_sizes; (void)n_in; (void)out_size; (void)ws_size;
  const float* x   = (const float*)d_in[0];
  const float* Wr  = (const float*)d_in[1];
  const float* Wg  = (const float*)d_in[2];
  const float* Wu  = (const float*)d_in[3];
  const float* Wd  = (const float*)d_in[4];
  const float* Wgs = (const float*)d_in[5];
  const float* Wus = (const float*)d_in[6];
  const float* Wds = (const float*)d_in[7];
  float* out    = (float*)d_out;
  float* logits = out + (size_t)T_TOK * DMODEL;

  char* ws = (char*)d_ws;
  size_t o = 0;
  auto alloc = [&](size_t bytes) { size_t r = o; o += (bytes + 255) & ~(size_t)255; return r; };
  u16*   xb      = (u16*)(ws + alloc((size_t)T_TOK * DMODEL * 2));
  u16*   Wg_pk   = (u16*)(ws + alloc((size_t)NEXP * DMODEL * ISZ * 2));
  u16*   Wu_pk   = (u16*)(ws + alloc((size_t)NEXP * DMODEL * ISZ * 2));
  u16*   Wd_pk   = (u16*)(ws + alloc((size_t)NEXP * ISZ * DMODEL * 2));
  u16*   Wgs_pk  = (u16*)(ws + alloc((size_t)DMODEL * SISZ * 2));
  u16*   Wus_pk  = (u16*)(ws + alloc((size_t)DMODEL * SISZ * 2));
  u16*   Wds_pk  = (u16*)(ws + alloc((size_t)SISZ * DMODEL * 2));
  u16*   h_r     = (u16*)(ws + alloc((size_t)CAP * ISZ * 2));
  u16*   h_s     = (u16*)(ws + alloc((size_t)T_TOK * SISZ * 2));
  float* y_r     = (float*)(ws + alloc((size_t)CAP * DMODEL * 4));
  int*   topi    = (int*)(ws + alloc((size_t)T_TOK * 2 * 4));
  float* topw    = (float*)(ws + alloc((size_t)T_TOK * 2 * 4));
  int*   tlist   = (int*)(ws + alloc((size_t)CAP * 4));
  float* slotw   = (float*)(ws + alloc((size_t)CAP * 4));
  int*   slotof  = (int*)(ws + alloc((size_t)T_TOK * 2 * 4));
  int*   meta    = (int*)(ws + alloc(32 * 4));

  const int BT = 256;
  // 1) convert + pack
  int nx = T_TOK * DMODEL;
  k_f32_to_bf16<<<(nx + BT - 1) / BT, BT, 0, stream>>>(x, xb, nx);
  int totE = NEXP * DMODEL * ISZ;
  k_pack_b<<<(totE + BT - 1) / BT, BT, 0, stream>>>(Wg, Wg_pk, DMODEL, ISZ, NEXP);
  k_pack_b<<<(totE + BT - 1) / BT, BT, 0, stream>>>(Wu, Wu_pk, DMODEL, ISZ, NEXP);
  k_pack_b<<<(totE + BT - 1) / BT, BT, 0, stream>>>(Wd, Wd_pk, ISZ, DMODEL, NEXP);
  int totS = DMODEL * SISZ;
  k_pack_b<<<(totS + BT - 1) / BT, BT, 0, stream>>>(Wgs, Wgs_pk, DMODEL, SISZ, 1);
  k_pack_b<<<(totS + BT - 1) / BT, BT, 0, stream>>>(Wus, Wus_pk, DMODEL, SISZ, 1);
  k_pack_b<<<(totS + BT - 1) / BT, BT, 0, stream>>>(Wds, Wds_pk, SISZ, DMODEL, 1);
  // 2) router + gather
  k_router<<<(T_TOK * 32) / BT, BT, 0, stream>>>(x, Wr, logits, topi, topw);
  k_count<<<1, BT, 0, stream>>>(topi, meta, tlist, slotw);
  k_scatter<<<(T_TOK + BT - 1) / BT, BT, 0, stream>>>(topi, topw, meta, tlist, slotw, slotof);
  // 3) routed experts: blocks = supers * n_groups
  k_moe_gateup<<<MSUPER_MAX * (ISZ / 32), BT, 0, stream>>>(xb, Wg_pk, Wu_pk, meta, tlist,
                                                           slotw, h_r);
  k_moe_down<<<MSUPER_MAX * (DMODEL / 64), BT, 0, stream>>>(h_r, Wd_pk, meta, y_r);
  // 4) shared expert + combine
  k_sh_gateup<<<(T_TOK / 128) * (SISZ / 32), BT, 0, stream>>>(xb, Wgs_pk, Wus_pk, h_s);
  k_sh_down_combine<<<(T_TOK / 128) * (DMODEL / 64), BT, 0, stream>>>(h_s, Wds_pk, y_r,
                                                                      slotof, out);
}
